// ShiftedWindowAttention_60387240182155
// MI455X (gfx1250) — matrix operand order repaired and checked by reading, not compiled, not run on hardware
//
#include <hip/hip_runtime.h>
#include <hip/hip_bf16.h>
#include <hip/hip_fp16.h>

typedef __attribute__((ext_vector_type(16))) _Float16 v16h;
typedef __attribute__((ext_vector_type(8)))  _Float16 v8h;
typedef __attribute__((ext_vector_type(8)))  float    v8f;
typedef __attribute__((ext_vector_type(4)))  unsigned int v4u;
typedef __attribute__((ext_vector_type(8)))  int      v8i;
typedef __attribute__((ext_vector_type(4)))  int      v4i;

#define WSZ    7
#define HEADS  8
#define SHFT   3
#define NB     4
#define NC     256
#define NH     224
#define NW     224
#define HD     32
#define HN     32
#define WN     32
#define WS2    49
#define NPIX   (NH * NW)          // 50176
#define XS_STRIDE 40              // LDS row stride (halves): 80B, 16B aligned
#define VT_STRIDE 72              // LDS row stride (halves): 144B, 16B aligned

// ---------------------------------------------------------------------------
// Kernel 0: convert the 768x256 weight matrix to f16 (one-time, tiny).
// ---------------------------------------------------------------------------
__global__ void cvt_w_kernel(const float* __restrict__ w,
                             _Float16* __restrict__ wh, int n) {
    int i = blockIdx.x * 256 + threadIdx.x;
    if (i < n) wh[i] = (_Float16)w[i];
}

// ---------------------------------------------------------------------------
// Kernel 1: QKV projection GEMM, fused shift-roll + window partition.
// Block (256 thr, 8 waves) computes 256(out-ch) x 64(pixel). Each K-step:
// coalesced-load 32ch x 64px of x -> f16 -> LDS transposed; each wave runs
// 8 WMMAs (32 rows x 64 cols; two A sub-tiles share four B tiles). Output
// scattered as f16 into [3, B, Hn, Wn, heads, 49, 32] in workspace.
// ---------------------------------------------------------------------------
__global__ __launch_bounds__(256)
void qkv_gemm_kernel(const float* __restrict__ x,
                     const _Float16* __restrict__ wh,
                     const float* __restrict__ bias,
                     _Float16* __restrict__ qkv) {
    __shared__ _Float16 Xs[64 * XS_STRIDE];

    const int tid  = threadIdx.x;
    const int lane = tid & 31;
    const int wave = tid >> 5;
    const int ln   = lane & 15;
    const int lg   = lane >> 4;

    const int n0 = blockIdx.x * 64;        // pixel base
    const int m0 = blockIdx.y * 256;       // out-channel base
    const int b  = blockIdx.z;

    // Staging address for this thread: pixel sp (0..63), channel group scl.
    const int sp  = tid & 63;
    const int scl = tid >> 6;              // 0..3
    const int pS  = n0 + sp;
    const int hS  = pS / NW, wS = pS % NW;
    const int hsS = (hS + SHFT) % NH;      // roll(-SHIFT) => source +SHIFT
    const int wsS = (wS + SHFT) % NW;
    const float* xb = x + (size_t)b * NC * NPIX + (size_t)hsS * NW + wsS;

    // A rows for this wave (f16 weights, K-contiguous in global).
    const int orow0 = m0 + wave * 32;
    const _Float16* wrow0 = wh + (size_t)(orow0 + ln) * NC;
    const _Float16* wrow1 = wh + (size_t)(orow0 + 16 + ln) * NC;

    v8f acc[2][4] = {};

    for (int k0 = 0; k0 < NC; k0 += 32) {
        __syncthreads();                   // previous-iter Xs reads done
        #pragma unroll
        for (int ps = 0; ps < 8; ++ps) {   // 8 passes x (4ch x 64px)
            int cl = ps * 4 + scl;         // local channel 0..31
            Xs[sp * XS_STRIDE + cl] = (_Float16)xb[(size_t)(k0 + cl) * NPIX];
        }
        __syncthreads();

        // Two A tiles: 16x32 f16, lane-group K slices {0-7,16-23}/{8-15,24-31}
        v16h a0, a1;
        {
            v8h lo = *(const v8h*)(wrow0 + k0 + lg * 8);
            v8h hi = *(const v8h*)(wrow0 + k0 + 16 + lg * 8);
            #pragma unroll
            for (int i = 0; i < 8; ++i) { a0[i] = lo[i]; a0[8 + i] = hi[i]; }
        }
        {
            v8h lo = *(const v8h*)(wrow1 + k0 + lg * 8);
            v8h hi = *(const v8h*)(wrow1 + k0 + 16 + lg * 8);
            #pragma unroll
            for (int i = 0; i < 8; ++i) { a1[i] = lo[i]; a1[8 + i] = hi[i]; }
        }

        #pragma unroll
        for (int nt = 0; nt < 4; ++nt) {
            int p = nt * 16 + ln;          // local pixel = B column
            v8h blo = *(const v8h*)(&Xs[p * XS_STRIDE + lg * 8]);
            v8h bhi = *(const v8h*)(&Xs[p * XS_STRIDE + 16 + lg * 8]);
            v16h bm;
            #pragma unroll
            for (int i = 0; i < 8; ++i) { bm[i] = blo[i]; bm[8 + i] = bhi[i]; }
            acc[0][nt] = __builtin_amdgcn_wmma_f32_16x16x32_f16(
                false, a0, false, bm, (short)0, acc[0][nt], false, false);
            acc[1][nt] = __builtin_amdgcn_wmma_f32_16x16x32_f16(
                false, a1, false, bm, (short)0, acc[1][nt], false, false);
        }
    }

    // Per-column window decode (independent of output row).
    int    qq[4];
    size_t wbase[4];
    #pragma unroll
    for (int nt = 0; nt < 4; ++nt) {
        int p  = n0 + nt * 16 + ln;
        int h  = p / NW, wc = p % NW;
        int wy = h / WSZ,  iy = h % WSZ;
        int wx = wc / WSZ, ix = wc % WSZ;
        qq[nt]    = iy * WSZ + ix;
        wbase[nt] = ((size_t)((b * HN + wy) * WN + wx)) * HEADS;
    }
    const size_t WHS = (size_t)NB * HN * WN * HEADS * (WS2 * HD);

    #pragma unroll
    for (int st = 0; st < 2; ++st) {
        #pragma unroll
        for (int r = 0; r < 8; ++r) {
            int o = orow0 + st * 16 + r + 8 * lg;   // 0..767
            float bv = bias[o];
            int qi = o >> 8, hh = (o >> 5) & 7, dd = o & 31;
            #pragma unroll
            for (int nt = 0; nt < 4; ++nt) {
                float v = acc[st][nt][r] + bv;
                size_t dst = (size_t)qi * WHS
                           + (wbase[nt] + hh) * (size_t)(WS2 * HD)
                           + (size_t)qq[nt] * HD + dd;
                qkv[dst] = (_Float16)v;
            }
        }
    }
}

// ---------------------------------------------------------------------------
// TDM helper: 1-D tensor copy global -> LDS (data_size = 2 bytes).
// Descriptor per CDNA5 ISA (D# group0/group1).
// Toolchain here is the 6-arg clang-23 form:
//   (uint32x4 g0, int32x8 g1, int32x4, int32x4, int32x8, i32 cpol)
// ---------------------------------------------------------------------------
__device__ __forceinline__ unsigned lds_off(const void* p) {
    // Generic pointers into LDS carry the offset in the low 32 bits.
    return (unsigned)(unsigned long long)p;
}

__device__ __forceinline__ void tdm_copy_1d(const void* gsrc, unsigned ldsDst,
                                            unsigned nelem) {
    unsigned long long ga = (unsigned long long)gsrc;
    v4u g0;
    g0[0] = 1u;                                  // count=1 (valid), user mode
    g0[1] = ldsDst;                              // lds_addr (bytes)
    g0[2] = (unsigned)(ga & 0xFFFFFFFFu);        // global_addr[31:0]
    g0[3] = (unsigned)((ga >> 32) & 0x01FFFFFFu) // global_addr[56:32]
          | (2u << 30);                          // type = 2 ("image")
    v8i g1;
    g1[0] = (int)(1u << 16);                     // data_size=1 -> 2 bytes
    g1[1] = (int)((nelem & 0xFFFFu) << 16);      // tensor_dim0[15:0]
    g1[2] = (int)(nelem >> 16);                  // tensor_dim0[31:16]
    g1[3] = (int)((nelem & 0xFFFFu) << 16);      // tile_dim0
    g1[4] = 0;                                   // tile_dim1=0 (1-D), tile_dim2=0
    g1[5] = (int)nelem;                          // tensor_dim0_stride[31:0]
    g1[6] = 0;
    g1[7] = 0;
    v4i z4 = {};
    v8i z8 = {};
    __builtin_amdgcn_tensor_load_to_lds(g0, g1, z4, z4, z8, 0);
}

// ---------------------------------------------------------------------------
// Kernel 2: per-window-head SDPA. 128 threads = 4 waves; wave m owns query
// rows [16m, 16m+16). Q/K staged via TDM tensor_load_to_lds; V staged
// transposed; scores via 4 WMMAs (K=32), masked softmax in registers,
// P -> LDS f16, out via 2x2 WMMAs, fused inverse roll on the scatter-store.
// ---------------------------------------------------------------------------
__global__ __launch_bounds__(128)
void win_attn_kernel(const _Float16* __restrict__ qkv,
                     float* __restrict__ out) {
    const int id   = blockIdx.x;       // ((b*Hn + wy)*Wn + wx)*heads + head
    const int head = id & 7;
    const int wx   = (id >> 3) & 31;
    const int wy   = (id >> 8) & 31;
    const int b    = id >> 13;

    __shared__ _Float16 Qs[64 * 32];
    __shared__ _Float16 Ks[64 * 32];
    __shared__ _Float16 Vt[32 * VT_STRIDE];  // transposed: Vt[d][pos]
    __shared__ _Float16 Ps[64 * 64];

    const size_t whStride = (size_t)NB * HN * WN * HEADS * (WS2 * HD);
    const size_t base     = (size_t)id * (WS2 * HD);
    const _Float16* Qg = qkv + base;
    const _Float16* Kg = qkv + whStride + base;
    const _Float16* Vg = qkv + 2 * whStride + base;

    // Async Q/K staging: one DMA each, issued by wave 0 only (TDM ignores
    // EXEC; the wave-uniform branch avoids 4x duplicate transfers).
    if (threadIdx.x < 32) {
        tdm_copy_1d(Qg, lds_off(&Qs[0]), WS2 * HD);
        tdm_copy_1d(Kg, lds_off(&Ks[0]), WS2 * HD);
        __builtin_amdgcn_s_wait_tensorcnt(0);
    }

    // V staged transposed (TDM cannot transpose); also zero-pad rows 49..63
    // of Qs/Ks (disjoint from the TDM-written rows 0..48).
    for (int j = threadIdx.x; j < 256; j += 128) {   // 64 rows x 4 segs
        int row = j >> 2, seg = j & 3;
        v8h v8 = {};
        if (row < WS2) {
            v8 = *(const v8h*)(Vg + row * HD + seg * 8);
        } else {
            v8h z = {};
            *(v8h*)(&Qs[row * 32 + seg * 8]) = z;
            *(v8h*)(&Ks[row * 32 + seg * 8]) = z;
        }
        #pragma unroll
        for (int e = 0; e < 8; ++e)
            Vt[(seg * 8 + e) * VT_STRIDE + row] = v8[e];
    }
    __syncthreads();

    const int lane = threadIdx.x & 31;
    const int wave = threadIdx.x >> 5;
    const int ln   = lane & 15;
    const int lg   = lane >> 4;

    // A tile: this wave's 16 query rows, K=32 (head dim), one WMMA k-step.
    const int qrow = wave * 16 + ln;
    v16h aq;
    {
        v8h lo = *(const v8h*)(&Qs[qrow * 32 + lg * 8]);
        v8h hi = *(const v8h*)(&Qs[qrow * 32 + 16 + lg * 8]);
        #pragma unroll
        for (int i = 0; i < 8; ++i) { aq[i] = lo[i]; aq[8 + i] = hi[i]; }
    }

    // Scores: 16 x 64 (4 N-tiles of key rows)
    v8f s[4];
    #pragma unroll
    for (int nt = 0; nt < 4; ++nt) {
        int krow = nt * 16 + ln;                   // key row = B column
        v8h lo = *(const v8h*)(&Ks[krow * 32 + lg * 8]);
        v8h hi = *(const v8h*)(&Ks[krow * 32 + 16 + lg * 8]);
        v16h bk;
        #pragma unroll
        for (int i = 0; i < 8; ++i) { bk[i] = lo[i]; bk[8 + i] = hi[i]; }
        v8f z = {};
        s[nt] = __builtin_amdgcn_wmma_f32_16x16x32_f16(
            false, aq, false, bk, (short)0, z, false, false);
    }

    // Masked softmax per row; C-layout row = r + 8*lg, cols striped on ln.
    const float SCALE = 0.17677669529663687f;      // 1/sqrt(32)
    #pragma unroll
    for (int r = 0; r < 8; ++r) {
        float pv[4];
        float mx = -1e30f;
        #pragma unroll
        for (int nt = 0; nt < 4; ++nt) {
            float val = s[nt][r] * SCALE;
            if (nt * 16 + ln >= WS2) val = -1e30f; // mask pad columns
            pv[nt] = val;
            mx = fmaxf(mx, val);
        }
        #pragma unroll
        for (int off = 1; off < 16; off <<= 1)
            mx = fmaxf(mx, __shfl_xor(mx, off, 32));
        float sum = 0.f;
        #pragma unroll
        for (int nt = 0; nt < 4; ++nt) {
            float e = __expf(pv[nt] - mx);
            pv[nt] = e;
            sum += e;
        }
        #pragma unroll
        for (int off = 1; off < 16; off <<= 1)
            sum += __shfl_xor(sum, off, 32);
        float inv = 1.0f / sum;
        int prow = wave * 16 + r + 8 * lg;
        #pragma unroll
        for (int nt = 0; nt < 4; ++nt)
            Ps[prow * 64 + nt * 16 + ln] = (_Float16)(pv[nt] * inv);
    }
    // Same-wave cross-lane LDS handoff (DS ops are in-order per wave).
    asm volatile("s_wait_dscnt 0" ::: "memory");

    // Out = P(16x64) x V(64x32): 2 K-steps x 2 N-tiles; V read from Vt.
    v8f o0 = {}, o1 = {};
    #pragma unroll
    for (int ks = 0; ks < 2; ++ks) {
        const int k0 = ks * 32;
        const int prow = wave * 16 + ln;
        v8h lo = *(const v8h*)(&Ps[prow * 64 + k0 + lg * 8]);
        v8h hi = *(const v8h*)(&Ps[prow * 64 + k0 + 16 + lg * 8]);
        v16h ap;
        #pragma unroll
        for (int i = 0; i < 8; ++i) { ap[i] = lo[i]; ap[8 + i] = hi[i]; }

        v8h b0lo = *(const v8h*)(&Vt[ln * VT_STRIDE + k0 + lg * 8]);
        v8h b0hi = *(const v8h*)(&Vt[ln * VT_STRIDE + k0 + 16 + lg * 8]);
        v8h b1lo = *(const v8h*)(&Vt[(16 + ln) * VT_STRIDE + k0 + lg * 8]);
        v8h b1hi = *(const v8h*)(&Vt[(16 + ln) * VT_STRIDE + k0 + 16 + lg * 8]);
        v16h bv0, bv1;
        #pragma unroll
        for (int i = 0; i < 8; ++i) {
            bv0[i] = b0lo[i]; bv0[8 + i] = b0hi[i];
            bv1[i] = b1lo[i]; bv1[8 + i] = b1hi[i];
        }
        o0 = __builtin_amdgcn_wmma_f32_16x16x32_f16(
            false, ap, false, bv0, (short)0, o0, false, false);
        o1 = __builtin_amdgcn_wmma_f32_16x16x32_f16(
            false, ap, false, bv1, (short)0, o1, false, false);
    }

    // Window merge + inverse roll (+SHIFT) scatter store.
    #pragma unroll
    for (int r = 0; r < 8; ++r) {
        int q = wave * 16 + r + 8 * lg;
        if (q < WS2) {
            int iy = q / WSZ, ix = q % WSZ;
            int h2 = (wy * WSZ + iy + SHFT) % NH;
            int w2 = (wx * WSZ + ix + SHFT) % NW;
            int ch0 = head * 32 + ln;
            int ch1 = head * 32 + 16 + ln;
            out[(((size_t)b * NC + ch0) * NH + h2) * NW + w2] = o0[r];
            out[(((size_t)b * NC + ch1) * NH + h2) * NW + w2] = o1[r];
        }
    }
}

// ---------------------------------------------------------------------------
extern "C" void kernel_launch(void* const* d_in, const int* in_sizes, int n_in,
                              void* d_out, int out_size, void* d_ws, size_t ws_size,
                              hipStream_t stream) {
    const float* x     = (const float*)d_in[0];
    const float* w_qkv = (const float*)d_in[1];
    const float* b_qkv = (const float*)d_in[2];
    float* out = (float*)d_out;

    // Workspace: [3,B,Hn,Wn,heads,49,32] f16 qkv, then f16 weights.
    _Float16* qkvh = (_Float16*)d_ws;
    const size_t qkvElems = (size_t)3 * NB * HN * WN * HEADS * WS2 * HD;
    _Float16* wh = qkvh + qkvElems;

    const int wn = 3 * NC * NC; // 768*256
    cvt_w_kernel<<<(wn + 255) / 256, 256, 0, stream>>>(w_qkv, wh, wn);

    dim3 g1(NPIX / 64, (3 * NC) / 256, NB);      // (784, 3, 4)
    qkv_gemm_kernel<<<g1, 256, 0, stream>>>(x, wh, b_qkv, qkvh);

    const int nWinHeads = NB * HN * WN * HEADS;  // 32768
    win_attn_kernel<<<nWinHeads, 128, 0, stream>>>(qkvh, out);
}
